// MultiBranchTransformerBlock_81415400063492
// MI455X (gfx1250) — matrix operand order repaired and checked by reading, not compiled, not run on hardware
//
#include <hip/hip_runtime.h>
#include <math.h>

#define B_   4
#define S_   2048
#define D_   768
#define NH_  12
#define HD_  64
#define E_   8
#define HID_ 3072
#define T_   (B_*S_)
#define TD_  (3*D_)
#define EPS_ 1.1920929e-07f

typedef __attribute__((ext_vector_type(16))) __bf16        bfvec16;
typedef __attribute__((ext_vector_type(8)))  float         v8f;
typedef __attribute__((ext_vector_type(4)))  unsigned int  u32x4;
typedef unsigned short u16;

// GCC-style int4 vector + address-space-qualified variants for the async-LDS builtin
typedef int v4i_t __attribute__((vector_size(16)));
typedef __attribute__((address_space(1))) v4i_t gas_v4i;
typedef __attribute__((address_space(3))) v4i_t las_v4i;

#if defined(__has_builtin)
#  if __has_builtin(__builtin_amdgcn_global_load_async_to_lds_b128)
#    define HAVE_ASYNC_LDS 1
#  else
#    define HAVE_ASYNC_LDS 0
#  endif
#  if __has_builtin(__builtin_amdgcn_update_dpp)
#    define HAVE_DPP 1
#  else
#    define HAVE_DPP 0
#  endif
#else
#  define HAVE_ASYNC_LDS 0
#  define HAVE_DPP 0
#endif

// ---------- bf16 helpers (RNE convert, storage as u16) ----------
__device__ __forceinline__ u16 f2bf(float f){
    unsigned u = __float_as_uint(f);
    u += 0x7FFFu + ((u >> 16) & 1u);
    return (u16)(u >> 16);
}
__device__ __forceinline__ float bf2f(u16 s){
    return __uint_as_float(((unsigned)s) << 16);
}

union FragU { bfvec16 v; u32x4 q[2]; };

// A-fragment (16x32 bf16): lane<16 holds K {0..7,16..23}, lane>=16 holds K {8..15,24..31}
__device__ __forceinline__ bfvec16 load_frag_a(const u16* p, int half){
    FragU r;
    r.q[0] = *(const u32x4*)(p + 8 * half);
    r.q[1] = *(const u32x4*)(p + 16 + 8 * half);
    return r.v;
}
// B-fragment (32x16 bf16): lane<16 holds K {0..7,16..23}, lane>=16 holds K {8..15,24..31}
__device__ __forceinline__ bfvec16 load_frag_b(const u16* p, int half){
    FragU r;
    r.q[0] = *(const u32x4*)(p + 8 * half);
    r.q[1] = *(const u32x4*)(p + 16 + 8 * half);
    return r.v;
}
__device__ __forceinline__ v8f wmma_bf16(bfvec16 a, bfvec16 b, v8f c){
    return __builtin_amdgcn_wmma_f32_16x16x32_bf16(false, a, false, b, (short)0, c, false, false);
}

// ---------- wave sync / async helpers ----------
template<int N> __device__ __forceinline__ void wait_async(){
#if HAVE_ASYNC_LDS
#  if __has_builtin(__builtin_amdgcn_s_wait_asynccnt)
    __builtin_amdgcn_s_wait_asynccnt((short)N);
#  else
    asm volatile("s_wait_asynccnt %0" :: "i"(N) : "memory");
#  endif
#endif
}

// ---------- DPP 16-lane butterfly reductions (pure VALU, no LDS) ----------
#if HAVE_DPP
template<int CTRL>
__device__ __forceinline__ float dpp_mov(float x){
    return __int_as_float(__builtin_amdgcn_update_dpp(0, __float_as_int(x), CTRL, 0xF, 0xF, true));
}
__device__ __forceinline__ float redmax16(float x){
    x = fmaxf(x, dpp_mov<0xB1>(x));   // quad_perm(1,0,3,2)  xor 1
    x = fmaxf(x, dpp_mov<0x4E>(x));   // quad_perm(2,3,0,1)  xor 2
    x = fmaxf(x, dpp_mov<0x141>(x));  // row_half_mirror     xor 4 (within 8)
    x = fmaxf(x, dpp_mov<0x140>(x));  // row_mirror          combine halves of 16
    return x;
}
__device__ __forceinline__ float redsum16(float x){
    x += dpp_mov<0xB1>(x);
    x += dpp_mov<0x4E>(x);
    x += dpp_mov<0x141>(x);
    x += dpp_mov<0x140>(x);
    return x;
}
#else
__device__ __forceinline__ float redmax16(float x){
    for (int o = 8; o; o >>= 1) x = fmaxf(x, __shfl_xor(x, o, 32));
    return x;
}
__device__ __forceinline__ float redsum16(float x){
    for (int o = 8; o; o >>= 1) x += __shfl_xor(x, o, 32);
    return x;
}
#endif

// ---------- elementwise weight prep ----------
__global__ void cvt_bf16_kernel(const float* __restrict__ in, u16* __restrict__ out, int n){
    int idx = blockIdx.x * 256 + threadIdx.x;
    if (idx < n) out[idx] = f2bf(in[idx]);
}

// in: [E][R][C] f32 -> out: [E][C][R] bf16
__global__ void transpose_cvt_kernel(const float* __restrict__ in, u16* __restrict__ out,
                                     int R, int C, int total){
    int idx = blockIdx.x * 256 + threadIdx.x;
    if (idx >= total) return;
    int rc  = R * C;
    int e   = idx / rc;
    int rem = idx - e * rc;
    int r   = rem / C;
    int c   = rem - r * C;
    out[(size_t)e * rc + (size_t)c * R + r] = f2bf(in[idx]);
}

// ---------- RMSNorm -> bf16 ----------
__global__ void rmsnorm_bf16_kernel(const float* __restrict__ x, const float* __restrict__ w,
                                    u16* __restrict__ out){
    int t = blockIdx.x, tid = threadIdx.x;
    __shared__ float red[256];
    const float* row = x + (size_t)t * D_;
    float ss = 0.f;
    for (int d = tid; d < D_; d += 256){ float v = row[d]; ss += v * v; }
    red[tid] = ss; __syncthreads();
    for (int o = 128; o > 0; o >>= 1){ if (tid < o) red[tid] += red[tid + o]; __syncthreads(); }
    float scale = rsqrtf(red[0] / (float)D_ + EPS_);
    for (int d = tid; d < D_; d += 256)
        out[(size_t)t * D_ + d] = f2bf(row[d] * scale * w[d]);
}

// ---------- B-tile stage: 64 rows x 32 k of bf16 (4KB) via async copy to LDS ----------
__device__ __forceinline__ void stage_b_tile(const u16* __restrict__ src, // Bt + n0*K + k0
                                             u16* dst, int K, int tid){
    int row  = tid >> 1;
    int part = (tid & 1) * 16;
    const u16* g0 = src + (size_t)row * K + part;
    u16* l0 = dst + row * 32 + part;
#if HAVE_ASYNC_LDS
    __builtin_amdgcn_global_load_async_to_lds_b128((gas_v4i*)g0,       (las_v4i*)l0,       0, 0);
    __builtin_amdgcn_global_load_async_to_lds_b128((gas_v4i*)(g0 + 8), (las_v4i*)(l0 + 8), 0, 0);
#else
    *(u32x4*)l0       = *(const u32x4*)g0;
    *(u32x4*)(l0 + 8) = *(const u32x4*)(g0 + 8);
#endif
}

// ---------- generic bf16 WMMA GEMM: C[M,N] = A[M,K] @ Bt[N,K]^T + bias ----------
// Block = 4 waves (128 thr), 64x64 output tile; B tile double-buffered in LDS (async copy).
// EPI: 0=store f32, 1=store bf16, 2=residual+store f32, 3=GELU store bf16, 4=rowscale accumulate f32
template<int EPI>
__global__ void gemm_bf16_kernel(const u16* __restrict__ A, const u16* __restrict__ Bt,
                                 const float* __restrict__ bias, int N, int K,
                                 float* __restrict__ Cf, u16* __restrict__ Cb,
                                 const float* __restrict__ res,
                                 const float* __restrict__ rowscale, int sstride){
    int tid = threadIdx.x;
    int lane = tid & 31, wv = tid >> 5, half = lane >> 4, lm = lane & 15;
    int m0 = blockIdx.y * 64 + wv * 16, n0 = blockIdx.x * 64;
    __shared__ u16 bstage[2][64 * 32];

    const u16* arow  = A  + (size_t)(m0 + lm) * K;
    const u16* bbase = Bt + (size_t)n0 * K;
    int nk = K >> 5;
    v8f acc[4] = {};

    stage_b_tile(bbase, bstage[0], K, tid);
    for (int kt = 0; kt < nk; ++kt){
        if (kt + 1 < nk){
            stage_b_tile(bbase + (kt + 1) * 32, bstage[(kt + 1) & 1], K, tid);
            wait_async<2>();                 // tile kt landed; tile kt+1 still in flight
            __builtin_prefetch(arow + (size_t)(kt + 2) * 32, 0, 1);
        } else {
            wait_async<0>();
        }
        __syncthreads();
        bfvec16 af = load_frag_a(arow + (size_t)kt * 32, half);
        const u16* bl = bstage[kt & 1];
        #pragma unroll
        for (int j = 0; j < 4; j++){
            acc[j] = wmma_bf16(af, load_frag_b(bl + (j * 16 + lm) * 32, half), acc[j]);
        }
        __syncthreads();                     // all waves done reading before overwrite
    }
    #pragma unroll
    for (int j = 0; j < 4; j++){
        #pragma unroll
        for (int r = 0; r < 8; r++){
            int m = m0 + r + 8 * half;
            int n = n0 + j * 16 + lm;
            size_t idx = (size_t)m * N + n;
            float v = acc[j][r] + bias[n];
            if      (EPI == 0) Cf[idx] = v;
            else if (EPI == 1) Cb[idx] = f2bf(v);
            else if (EPI == 2) Cf[idx] = res[idx] + v;
            else if (EPI == 3){ float g = 0.5f * v * (1.f + erff(v * 0.70710678118f)); Cb[idx] = f2bf(g); }
            else               Cf[idx] += rowscale[(size_t)m * sstride] * v;
        }
    }
}

// ---------- flash attention: one wave per (b, h, 16 q-rows) ----------
__global__ void attention_kernel(const u16* __restrict__ qkv, u16* __restrict__ ctx){
    const int ROW = TD_;
    int q0 = blockIdx.x * 16, h = blockIdx.y, b = blockIdx.z;
    int lane = threadIdx.x, half = lane >> 4, lm = lane & 15;
    __shared__ u16 lds_p[16 * 32];
    __shared__ u16 lds_vt[64 * 32];
    size_t base = (size_t)b * S_ * ROW;

    const u16* qrow = qkv + base + (size_t)(q0 + lm) * ROW + h * HD_;
    bfvec16 qa0 = load_frag_a(qrow, half);         // hd 0..31
    bfvec16 qa1 = load_frag_a(qrow + 32, half);    // hd 32..63

    float mrow[8], lrow[8];
    v8f o0 = {}, o1 = {}, o2 = {}, o3 = {};
    #pragma unroll
    for (int r = 0; r < 8; r++){ mrow[r] = -INFINITY; lrow[r] = 0.f; }

    for (int k0 = 0; k0 < S_; k0 += 32){
        const u16* krowA = qkv + base + (size_t)(k0 + lm) * ROW + D_ + h * HD_;
        const u16* krowB = krowA + (size_t)16 * ROW;
        v8f c0 = {}, c1 = {};
        c0 = wmma_bf16(qa0, load_frag_b(krowA,      half), c0);
        c0 = wmma_bf16(qa1, load_frag_b(krowA + 32, half), c0);
        c1 = wmma_bf16(qa0, load_frag_b(krowB,      half), c1);
        c1 = wmma_bf16(qa1, load_frag_b(krowB + 32, half), c1);

        float corr[8];
        #pragma unroll
        for (int r = 0; r < 8; r++){
            float s0 = c0[r] * 0.125f, s1 = c1[r] * 0.125f;
            float mn = fmaxf(mrow[r], redmax16(fmaxf(s0, s1)));
            corr[r]  = expf(mrow[r] - mn);
            float p0 = expf(s0 - mn), p1 = expf(s1 - mn);
            float rs = redsum16(p0 + p1);
            lrow[r] = lrow[r] * corr[r] + rs;
            mrow[r] = mn;
            lds_p[(r + 8 * half) * 32 + lm]      = f2bf(p0);
            lds_p[(r + 8 * half) * 32 + 16 + lm] = f2bf(p1);
        }
        // stage V transposed: lds_vt[hd][kv]
        {
            const u16* vrow = qkv + base + (size_t)(k0 + lane) * ROW + 2 * D_ + h * HD_;
            u32x4 vv[8];
            #pragma unroll
            for (int i = 0; i < 8; i++) vv[i] = ((const u32x4*)vrow)[i];
            #pragma unroll
            for (int i = 0; i < 8; i++){
                #pragma unroll
                for (int jj = 0; jj < 4; jj++){
                    unsigned wvv = vv[i][jj];
                    int d = i * 8 + jj * 2;
                    lds_vt[d * 32 + lane]       = (u16)(wvv & 0xFFFFu);
                    lds_vt[(d + 1) * 32 + lane] = (u16)(wvv >> 16);
                }
            }
        }
        __syncthreads();
        #pragma unroll
        for (int r = 0; r < 8; r++){
            o0[r] *= corr[r]; o1[r] *= corr[r]; o2[r] *= corr[r]; o3[r] *= corr[r];
        }
        bfvec16 pa = load_frag_a(&lds_p[lm * 32], half);
        o0 = wmma_bf16(pa, load_frag_b(&lds_vt[(0 * 16 + lm) * 32], half), o0);
        o1 = wmma_bf16(pa, load_frag_b(&lds_vt[(1 * 16 + lm) * 32], half), o1);
        o2 = wmma_bf16(pa, load_frag_b(&lds_vt[(2 * 16 + lm) * 32], half), o2);
        o3 = wmma_bf16(pa, load_frag_b(&lds_vt[(3 * 16 + lm) * 32], half), o3);
        __syncthreads();
    }
    #pragma unroll
    for (int r = 0; r < 8; r++){
        float inv = 1.f / lrow[r];
        size_t orow = ((size_t)b * S_ + q0 + r + 8 * half) * D_ + h * HD_ + lm;
        ctx[orow +  0] = f2bf(o0[r] * inv);
        ctx[orow + 16] = f2bf(o1[r] * inv);
        ctx[orow + 32] = f2bf(o2[r] * inv);
        ctx[orow + 48] = f2bf(o3[r] * inv);
    }
}

// ---------- router: one wave per token ----------
__global__ void router_kernel(const u16* __restrict__ hf, const float* __restrict__ rw,
                              const float* __restrict__ rb,
                              float* __restrict__ probs, float* __restrict__ comb){
    int lane = threadIdx.x & 31;
    int t = blockIdx.x * 8 + (threadIdx.x >> 5);
    const u16* row = hf + (size_t)t * D_;
    float acc[E_];
    #pragma unroll
    for (int e = 0; e < E_; e++) acc[e] = 0.f;
    for (int d = lane; d < D_; d += 32){
        float hv = bf2f(row[d]);
        #pragma unroll
        for (int e = 0; e < E_; e++) acc[e] += hv * rw[e * D_ + d];
    }
    #pragma unroll
    for (int e = 0; e < E_; e++)
        for (int o2 = 16; o2; o2 >>= 1) acc[e] += __shfl_xor(acc[e], o2, 32);
    if (lane == 0){
        float p[E_]; float mx = -1e30f;
        for (int e = 0; e < E_; e++){ p[e] = acc[e] + rb[e]; mx = fmaxf(mx, p[e]); }
        float sum = 0.f;
        for (int e = 0; e < E_; e++){ p[e] = expf(p[e] - mx); sum += p[e]; }
        float inv = 1.f / sum;
        for (int e = 0; e < E_; e++){ p[e] *= inv; probs[(size_t)t * E_ + e] = p[e]; }
        int i1 = 0;
        for (int e = 1; e < E_; e++) if (p[e] > p[i1]) i1 = e;
        int i2 = (i1 == 0) ? 1 : 0;
        for (int e = 0; e < E_; e++) if (e != i1 && p[e] > p[i2]) i2 = e;
        float s = p[i1] + p[i2];
        for (int e = 0; e < E_; e++)
            comb[(size_t)t * E_ + e] = (e == i1) ? p[i1] / s : ((e == i2) ? p[i2] / s : 0.f);
    }
}

// ---------- load-balance loss (deterministic two-pass) ----------
__global__ void loss_kernel(const float* __restrict__ probs, float* __restrict__ out){
    __shared__ float red[256];
    int tid = threadIdx.x;
    float loss = 0.f;
    for (int e = 0; e < E_; e++){
        float s = 0.f;
        for (int t = tid; t < T_; t += 256) s += probs[(size_t)t * E_ + e];
        red[tid] = s; __syncthreads();
        for (int o = 128; o > 0; o >>= 1){ if (tid < o) red[tid] += red[tid + o]; __syncthreads(); }
        if (tid == 0){ float pm = red[0] / (float)T_; loss += pm * logf(pm); }
        __syncthreads();
    }
    if (tid == 0) out[0] = (float)E_ * loss;
}

extern "C" void kernel_launch(void* const* d_in, const int* in_sizes, int n_in,
                              void* d_out, int out_size, void* d_ws, size_t ws_size,
                              hipStream_t stream){
    (void)in_sizes; (void)n_in; (void)out_size; (void)ws_size;
    const float* x    = (const float*)d_in[0];
    const float* anw  = (const float*)d_in[1];
    const float* wqkv = (const float*)d_in[2];
    const float* bqkv = (const float*)d_in[3];
    const float* wout = (const float*)d_in[4];
    const float* bout = (const float*)d_in[5];
    const float* fnw  = (const float*)d_in[6];
    const float* rw   = (const float*)d_in[7];
    const float* rb   = (const float*)d_in[8];
    const float* w1   = (const float*)d_in[9];
    const float* b1   = (const float*)d_in[10];
    const float* w2   = (const float*)d_in[11];
    const float* b2   = (const float*)d_in[12];
    float* out = (float*)d_out;

    char* ws = (char*)d_ws;
    size_t off = 0;
    auto alloc = [&](size_t bytes)->char*{
        char* p = ws + off;
        off = (off + bytes + 255) & ~(size_t)255;
        return p;
    };
    u16*   hbf    = (u16*)  alloc((size_t)T_ * D_ * 2);
    u16*   qkvbf  = (u16*)  alloc((size_t)T_ * TD_ * 2);
    u16*   ctxbf  = (u16*)  alloc((size_t)T_ * D_ * 2);
    float* x1     = (float*)alloc((size_t)T_ * D_ * 4);
    u16*   hfbf   = (u16*)  alloc((size_t)T_ * D_ * 2);
    u16*   wqkvbf = (u16*)  alloc((size_t)TD_ * D_ * 2);
    u16*   woutbf = (u16*)  alloc((size_t)D_ * D_ * 2);
    u16*   w1t    = (u16*)  alloc((size_t)E_ * D_ * HID_ * 2);
    u16*   w2t    = (u16*)  alloc((size_t)E_ * D_ * HID_ * 2);
    u16*   actbf  = (u16*)  alloc((size_t)T_ * HID_ * 2);
    float* probs  = (float*)alloc((size_t)T_ * E_ * 4);
    float* comb   = (float*)alloc((size_t)T_ * E_ * 4);

    // weight prep
    { int n = TD_ * D_;        cvt_bf16_kernel<<<(n + 255) / 256, 256, 0, stream>>>(wqkv, wqkvbf, n); }
    { int n = D_ * D_;         cvt_bf16_kernel<<<(n + 255) / 256, 256, 0, stream>>>(wout, woutbf, n); }
    { int n = E_ * D_ * HID_;  transpose_cvt_kernel<<<(n + 255) / 256, 256, 0, stream>>>(w1, w1t, D_, HID_, n); }
    { int n = E_ * HID_ * D_;  transpose_cvt_kernel<<<(n + 255) / 256, 256, 0, stream>>>(w2, w2t, HID_, D_, n); }

    // attention branch
    rmsnorm_bf16_kernel<<<T_, 256, 0, stream>>>(x, anw, hbf);
    gemm_bf16_kernel<1><<<dim3(TD_ / 64, T_ / 64), 128, 0, stream>>>(
        hbf, wqkvbf, bqkv, TD_, D_, nullptr, qkvbf, nullptr, nullptr, 0);
    attention_kernel<<<dim3(S_ / 16, NH_, B_), 32, 0, stream>>>(qkvbf, ctxbf);
    gemm_bf16_kernel<2><<<dim3(D_ / 64, T_ / 64), 128, 0, stream>>>(
        ctxbf, woutbf, bout, D_, D_, x1, nullptr, x, nullptr, 0);

    // residual into output, then MoE accumulates on top
    (void)hipMemcpyAsync(out, x1, (size_t)T_ * D_ * 4, hipMemcpyDeviceToDevice, stream);

    rmsnorm_bf16_kernel<<<T_, 256, 0, stream>>>(x1, fnw, hfbf);
    router_kernel<<<T_ / 8, 256, 0, stream>>>(hfbf, rw, rb, probs, comb);
    loss_kernel<<<1, 256, 0, stream>>>(probs, out + (size_t)T_ * D_);

    for (int e = 0; e < E_; e++){
        gemm_bf16_kernel<3><<<dim3(HID_ / 64, T_ / 64), 128, 0, stream>>>(
            hfbf, w1t + (size_t)e * D_ * HID_, b1 + e * HID_, HID_, D_,
            nullptr, actbf, nullptr, nullptr, 0);
        gemm_bf16_kernel<4><<<dim3(D_ / 64, T_ / 64), 128, 0, stream>>>(
            actbf, w2t + (size_t)e * D_ * HID_, b2 + e * D_, D_, HID_,
            out, nullptr, nullptr, comb + e, E_);
    }
}
